// LinearAttention_27951647163012
// MI455X (gfx1250) — matrix operand order repaired and checked by reading, not compiled, not run on hardware
//
#include <hip/hip_runtime.h>

// ---------------------------------------------------------------------------
// Problem constants (from reference)
// ---------------------------------------------------------------------------
static constexpr int B_BATCH = 2;
static constexpr int S_SEQ   = 2048;
static constexpr int F_DIM   = 512;
static constexpr int I_DIM   = 1024;
static constexpr int E_EXP   = 8;
static constexpr int K_TAP   = 7;
static constexpr int D_LAY   = 2;
static constexpr int N_CLS   = 8192;
static constexpr int F3      = 3 * F_DIM;        // 1536
static constexpr int T_TOK   = B_BATCH * S_SEQ;  // 4096

typedef unsigned int   u32;
typedef unsigned short u16;
typedef __attribute__((ext_vector_type(4)))  u32   u32x4;
typedef __attribute__((ext_vector_type(16))) __bf16 v16bf;
typedef __attribute__((ext_vector_type(8)))  float  v8f;

#define DEV __device__ __forceinline__

DEV u16 f2bf(float f) {               // f32 -> bf16 round-to-nearest-even
  u32 u = __builtin_bit_cast(u32, f);
  u += 0x7fffu + ((u >> 16) & 1u);
  return (u16)(u >> 16);
}

union FragBF { v16bf v; u32x4 u[2]; };

// ---------------------------------------------------------------------------
// WMMA GEMM: C(MxN) = A(MxK) * B(KxN); A and B are pre-converted bf16, f32
// accumulate via v_wmma_f32_16x16x32_bf16.  Inner loop is pure data movement
// (global_load_b128 -> ds_store_b128 staging, ds_load_b128 fragments) + WMMA.
// Block tile 128x64, 8 wave32 waves (4x2), each wave 2x2 16x16 accumulators.
// ---------------------------------------------------------------------------
enum { A_DIRECT = 0, A_GATHER = 1 };
enum { B_BF16ROW = 0, B_XPOSE = 1 };

static constexpr int TBM = 128, TBN = 64, TBK = 32, APAD = 8, BPAD = 8;

template <int AM, int BMO>
__global__ __launch_bounds__(256)
void gemm_bf16(const u16* __restrict__ A, int lda, int M, int Kdim,
               const u16* __restrict__ Bp, const u16* __restrict__ Bp2,
               int ldb, int bsplit,
               float* __restrict__ C, int ldc, u16* __restrict__ Cbf,
               int convI, int seqS,
               const int* __restrict__ glist, const int* __restrict__ goff,
               const int* __restrict__ gcnt, long bExpStride,
               const float* __restrict__ biasM, int reluStore, long cBatchStride)
{
  // LDS staged bf16 (ushort); padding keeps 16B fragment reads conflict-light
  // (A row stride 80B, B row stride 144B, both 16B aligned).
  __shared__ __align__(16) u16 As[2][TBM][TBK + APAD];
  __shared__ __align__(16) u16 Bs[2][TBK][TBN + BPAD];

  const int tid = threadIdx.x;
  const int n0  = blockIdx.x * TBN;
  const int m0  = blockIdx.y * TBM;

  int validM = M;
  const u16* Bb = Bp;
  float* Cb = C;
  const int* list = nullptr;
  if constexpr (AM == A_GATHER) {
    const int e   = blockIdx.z;
    const int cnt = gcnt[e];
    if (m0 >= cnt) return;                 // uniform early exit (before barriers)
    validM = cnt;
    const int off = goff[e];
    list = glist + off;
    Bb = Bp + (long)e * bExpStride;
    Cb = C + (long)off * ldc;
  }

  // staging thread mapping (256 threads)
  const int ar  = tid >> 1;            // A row 0..127
  const int ac  = (tid & 1) * 16;      // A col segment 0/16 (elements)
  const int br  = tid >> 3;            // B row 0..31 (row-major mode)
  const int bc  = (tid & 7) * 8;       // B col segment
  const int xn  = tid & 63;            // XPOSE: token column
  const int xr0 = (tid >> 6) * 8;      // XPOSE: k-row base

  u32x4 aReg[2];
  u32x4 bReg;

  const int ntap = convI ? (Kdim / convI) : 1;

  auto loadG = [&](int k0) {
    // ---- A tile (128 x 32) bf16, per-row gather / causal shift ----
    int shift = 0, acol = k0;
    if (convI) { int tap = k0 / convI; shift = (ntap - 1) - tap; acol = k0 - tap * convI; }
    long srow = -1;
    const int gm = m0 + ar;
    if constexpr (AM == A_GATHER) {
      if (gm < validM) srow = list[gm];
    } else {
      if (gm < M) {
        if (convI) { int s = gm % seqS; if (s >= shift) srow = gm - shift; }
        else       srow = gm;
      }
    }
    if (srow >= 0) {
      const u32x4* p = (const u32x4*)(A + srow * (long)lda + acol + ac);
      aReg[0] = p[0]; aReg[1] = p[1];
    } else {
      u32x4 z = {0u, 0u, 0u, 0u};
      aReg[0] = z; aReg[1] = z;
    }
    // ---- B tile (32 x 64) bf16 ----
    if constexpr (BMO == B_BF16ROW) {
      bReg = *(const u32x4*)(Bb + (long)(k0 + br) * ldb + n0 + bc);
    } else {  // B_XPOSE: B[k][n] = act_bf16[token n][channel k], x0|x1 split
      int ch = k0 + xr0;
      const u16* src = Bp;
      if (ch >= bsplit) { src = Bp2; ch -= bsplit; }
      bReg = *(const u32x4*)(src + (long)(n0 + xn) * ldb + ch);
    }
  };

  auto storeS = [&](int buf) {
    *(u32x4*)&As[buf][ar][ac]     = aReg[0];
    *(u32x4*)&As[buf][ar][ac + 8] = aReg[1];
    if constexpr (BMO == B_BF16ROW) {
      *(u32x4*)&Bs[buf][br][bc] = bReg;
    } else {
      union { u32x4 v; u16 s[8]; } bu; bu.v = bReg;
#pragma unroll
      for (int j = 0; j < 8; ++j) Bs[buf][xr0 + j][xn] = bu.s[j];
    }
  };

  // wave tiling
  const int lane = tid & 31;
  const int w    = tid >> 5;
  const int wm   = (w >> 1) * 32;
  const int wn   = (w & 1) * 32;
  const int lrow = lane & 15;
  const int hi   = lane >> 4;          // 0/1
  const int ac0  = hi ? 8 : 0;         // A-fragment K chunks {0,16}/{8,24}

  v8f acc[2][2] = {};

  const int nkt = Kdim / TBK;
  loadG(0);
  storeS(0);
  __syncthreads();
  int buf = 0;
  for (int kt = 0; kt < nkt; ++kt) {
    const bool more = (kt + 1 < nkt);
    if (more) loadG((kt + 1) * TBK);    // prefetch next tile into registers

    FragBF af[2], bfr[2];
#pragma unroll
    for (int mi = 0; mi < 2; ++mi) {
      const int row = wm + mi * 16 + lrow;
      af[mi].u[0] = *(const u32x4*)&As[buf][row][ac0];
      af[mi].u[1] = *(const u32x4*)&As[buf][row][ac0 + 16];
    }
#pragma unroll
    for (int ni = 0; ni < 2; ++ni) {
      const int col = wn + ni * 16;
#pragma unroll
      for (int s = 0; s < 16; ++s) {
        const int kk = (s < 8) ? (ac0 + s) : (16 + ac0 + (s - 8));
        bfr[ni].v[s] = __builtin_bit_cast(__bf16, Bs[buf][kk][col + lrow]);
      }
    }
#pragma unroll
    for (int mi = 0; mi < 2; ++mi)
#pragma unroll
      for (int ni = 0; ni < 2; ++ni)
        acc[mi][ni] = __builtin_amdgcn_wmma_f32_16x16x32_bf16(
            false, af[mi].v, false, bfr[ni].v, (short)0, acc[mi][ni], false, false);

    __syncthreads();
    if (more) { storeS(buf ^ 1); buf ^= 1; __syncthreads(); }
  }

  // ---- epilogue: C layout per WMMA f32 C/D map (VGPR r -> M=r / 8+r) ----
  float* Cout = Cb;
  long nbase = n0;
  if (cBatchStride) {                  // (B, CLASSES, S) output remap
    const int b = n0 / seqS;
    Cout  = Cb + (long)b * cBatchStride;
    nbase = n0 - (long)b * seqS;
  }
  const int ncl = lane & 15;
#pragma unroll
  for (int mi = 0; mi < 2; ++mi) {
#pragma unroll
    for (int ni = 0; ni < 2; ++ni) {
      const long ncol = nbase + wn + ni * 16 + ncl;
      const int  mt   = wm + mi * 16 + hi * 8;
#pragma unroll
      for (int r = 0; r < 8; ++r) {
        const int ml = mt + r;
        if (m0 + ml < validM) {
          float v = acc[mi][ni][r];
          if (biasM) v += biasM[m0 + ml];
          if (reluStore) v = fmaxf(v, 0.f);
          Cout[(long)(m0 + ml) * ldc + ncol] = v;
          if (Cbf) Cbf[(long)(m0 + ml) * ldc + ncol] = f2bf(v);  // bf16 mirror
        }
      }
    }
  }
}

// ---------------------------------------------------------------------------
// Small kernels
// ---------------------------------------------------------------------------
__global__ void k_cvt(const float* __restrict__ in, u16* __restrict__ out)
{
  long idx = (long)blockIdx.x * blockDim.x + threadIdx.x;
  out[idx] = f2bf(in[idx]);
}

__global__ void k_embed(const int* __restrict__ inp, const float* __restrict__ embed,
                        float* __restrict__ x0, float* __restrict__ x1,
                        u16* __restrict__ x0b, u16* __restrict__ x1b)
{
  long idx = (long)blockIdx.x * blockDim.x + threadIdx.x;  // over T*F
  int t = (int)(idx >> 9);
  int f = (int)(idx & 511);
  long e = (long)inp[t] * (2 * F_DIM);
  float a = embed[e + f], b = embed[e + F_DIM + f];
  x0[idx] = a;  x1[idx] = b;
  x0b[idx] = f2bf(a); x1b[idx] = f2bf(b);
}

__global__ void k_gate(const float* __restrict__ X, int Fin, const float* __restrict__ gw,
                       int* __restrict__ eidx, float* __restrict__ esc)
{
  int t = blockIdx.x * blockDim.x + threadIdx.x;
  const float* xr = X + (long)t * Fin;
  float a[E_EXP];
#pragma unroll
  for (int e = 0; e < E_EXP; ++e) a[e] = 0.f;
  for (int f = 0; f < Fin; ++f) {
    float xv = xr[f];
    const float* gr = gw + (long)f * E_EXP;
#pragma unroll
    for (int e = 0; e < E_EXP; ++e) a[e] = fmaf(xv, gr[e], a[e]);
  }
  int i0 = 0; float v0 = a[0];
#pragma unroll
  for (int e = 1; e < E_EXP; ++e) if (a[e] > v0) { v0 = a[e]; i0 = e; }
  int i1 = -1; float v1 = -3.4e38f;
#pragma unroll
  for (int e = 0; e < E_EXP; ++e) if (e != i0 && a[e] > v1) { v1 = a[e]; i1 = e; }
  float s0 = 1.f / (1.f + __expf(v1 - v0));   // softmax over top-2
  eidx[2 * t] = i0; eidx[2 * t + 1] = i1;
  esc[2 * t] = s0;  esc[2 * t + 1] = 1.f - s0;
}

__global__ void k_zero64(int* p) { p[threadIdx.x] = 0; }

__global__ void k_count(const int* __restrict__ eidx, int* __restrict__ cnt)
{
  int t = blockIdx.x * blockDim.x + threadIdx.x;
  atomicAdd(&cnt[eidx[2 * t]], 1);
  atomicAdd(&cnt[eidx[2 * t + 1]], 1);
}

__global__ void k_scan(const int* __restrict__ cnt, int* __restrict__ offs)
{
  offs[0] = 0;
  for (int e = 0; e < E_EXP; ++e) offs[e + 1] = offs[e] + cnt[e];
}

__global__ void k_bucket(const int* __restrict__ eidx, const int* __restrict__ offs,
                         int* __restrict__ cnt2, int* __restrict__ list,
                         int* __restrict__ posr)
{
  int t = blockIdx.x * blockDim.x + threadIdx.x;
#pragma unroll
  for (int j = 0; j < 2; ++j) {
    int e  = eidx[2 * t + j];
    int p  = atomicAdd(&cnt2[e], 1);
    int gp = offs[e] + p;
    list[gp] = t;
    posr[2 * t + j] = gp;   // deterministic gather position (order-independent)
  }
}

// top-2 weighted combine -> relu -> bf16 (feeds conv GEMM A)
__global__ void k_combine_bf(const float* __restrict__ Y, int No,
                             const int* __restrict__ posr, const float* __restrict__ esc,
                             u16* __restrict__ outp)
{
  long idx = (long)blockIdx.x * blockDim.x + threadIdx.x;
  int t = (int)(idx / No);
  int i = (int)(idx - (long)t * No);
  float v = esc[2 * t]     * Y[(long)posr[2 * t]     * No + i]
          + esc[2 * t + 1] * Y[(long)posr[2 * t + 1] * No + i];
  outp[idx] = f2bf(fmaxf(v, 0.f));
}

// top-2 weighted combine -> f32 (feeds cumsum)
__global__ void k_combine_f32(const float* __restrict__ Y, int No,
                              const int* __restrict__ posr, const float* __restrict__ esc,
                              float* __restrict__ outp)
{
  long idx = (long)blockIdx.x * blockDim.x + threadIdx.x;
  int t = (int)(idx / No);
  int i = (int)(idx - (long)t * No);
  outp[idx] = esc[2 * t]     * Y[(long)posr[2 * t]     * No + i]
            + esc[2 * t + 1] * Y[(long)posr[2 * t + 1] * No + i];
}

// conv_w (O,I,K) f32 -> bf16 (K,I,O) so the conv GEMM B loads are contiguous
__global__ void k_convw(const float* __restrict__ cw, u16* __restrict__ wt)
{
  long idx = (long)blockIdx.x * blockDim.x + threadIdx.x;  // over K*I*I
  int  o   = (int)(idx & (I_DIM - 1));
  long r   = idx >> 10;
  int  i   = (int)(r & (I_DIM - 1));
  int  tap = (int)(r >> 10);
  wt[idx] = f2bf(cw[((long)o * I_DIM + i) * K_TAP + tap]);
}

// o (T,1536) -> g (T,512): cumsum/divisor * scale + shift, one column per thread
__global__ void k_cumsum(const float* __restrict__ o, float* __restrict__ g)
{
  int id = blockIdx.x * blockDim.x + threadIdx.x;  // 0..B*F-1
  int b = id >> 9;
  int f = id & 511;
  float cum = 0.f;
  for (int s = 0; s < S_SEQ; ++s) {
    long t = (long)b * S_SEQ + s;
    const float* row = o + t * F3;
    cum += row[f];
    g[t * F_DIM + f] = cum / (float)(s + 1) * row[F_DIM + f] + row[2 * F_DIM + f];
  }
}

// per-token channel norm + residual recurrence (f32 state + bf16 mirrors):
// x0' = 0.9*x0 + 0.1*f ; x1' = x1 + x0'
__global__ void k_normupd(const float* __restrict__ g,
                          float* __restrict__ x0, float* __restrict__ x1,
                          u16* __restrict__ x0b, u16* __restrict__ x1b)
{
  __shared__ float sd0[256], sd1[256];
  int t = blockIdx.x, tid = threadIdx.x;
  const float* row = g + (long)t * F_DIM;
  float v0 = row[tid], v1 = row[tid + 256];
  sd0[tid] = v0 + v1;
  sd1[tid] = v0 * v0 + v1 * v1;
  __syncthreads();
  for (int st = 128; st > 0; st >>= 1) {
    if (tid < st) { sd0[tid] += sd0[tid + st]; sd1[tid] += sd1[tid + st]; }
    __syncthreads();
  }
  float mean = sd0[0] * (1.f / 512.f);
  float ss   = fmaxf(sd1[0] - 512.f * mean * mean, 0.f);
  float denom = sqrtf(ss) * 0.04419417382415922f + 1e-5f;  // ||.|| * 512^-0.5 + eps
  float inv   = 0.7071067811865476f / denom;               // INIT_SCALE / denom
  long base = (long)t * F_DIM;
#pragma unroll
  for (int half = 0; half < 2; ++half) {
    int j = tid + half * 256;
    float fv = ((half ? v1 : v0) - mean) * inv;
    float a0 = x0[base + j], a1 = x1[base + j];
    float nx0 = 0.9f * a0 + 0.1f * fv;
    float nx1 = a1 + nx0;
    x0[base + j] = nx0;  x1[base + j] = nx1;
    x0b[base + j] = f2bf(nx0);  x1b[base + j] = f2bf(nx1);
  }
}

// ---------------------------------------------------------------------------
// Host launcher
// ---------------------------------------------------------------------------
extern "C" void kernel_launch(void* const* d_in, const int* in_sizes, int n_in,
                              void* d_out, int out_size, void* d_ws, size_t ws_size,
                              hipStream_t stream)
{
  (void)in_sizes; (void)n_in; (void)out_size; (void)ws_size;
  const int*   inp     = (const int*)d_in[0];
  const float* embed   = (const float*)d_in[1];
  const float* g_in_w  = (const float*)d_in[2];
  const float* w_in    = (const float*)d_in[3];
  const float* conv_w  = (const float*)d_in[4];
  const float* g_out_w = (const float*)d_in[5];
  const float* w_out   = (const float*)d_in[6];
  const float* out_w   = (const float*)d_in[7];
  const float* out_b   = (const float*)d_in[8];
  float* logits = (float*)d_out;

  char* base = (char*)d_ws;
  size_t cur = 0;
  auto alloc = [&](size_t bytes) -> void* {
    void* p = base + cur;
    cur += (bytes + 255) & ~(size_t)255;
    return p;
  };

  float* x0    = (float*)alloc((size_t)T_TOK * F_DIM * 4);
  float* x1    = (float*)alloc((size_t)T_TOK * F_DIM * 4);
  u16*   x0b   = (u16*)  alloc((size_t)T_TOK * F_DIM * 2);
  u16*   x1b   = (u16*)  alloc((size_t)T_TOK * F_DIM * 2);
  u16*   h_bf  = (u16*)  alloc((size_t)T_TOK * I_DIM * 2);
  float* hc    = (float*)alloc((size_t)T_TOK * I_DIM * 4);
  u16*   hc_bf = (u16*)  alloc((size_t)T_TOK * I_DIM * 2);
  float* obuf  = (float*)alloc((size_t)T_TOK * F3 * 4);
  float* Y     = (float*)alloc((size_t)2 * T_TOK * F3 * 4);   // expert GEMM outputs
  float* g     = Y;   // alias: Y dead by the time cumsum writes g
  u16*   cwt   = (u16*)  alloc((size_t)K_TAP * I_DIM * I_DIM * 2);
  u16*   winb  = (u16*)  alloc((size_t)E_EXP * F_DIM * I_DIM * 2);
  u16*   woutb = (u16*)  alloc((size_t)E_EXP * I_DIM * F3 * 2);
  u16*   outwb = (u16*)  alloc((size_t)N_CLS * 2 * F_DIM * 2);
  int*   eidx  = (int*)  alloc((size_t)2 * T_TOK * 4);
  float* esc   = (float*)alloc((size_t)2 * T_TOK * 4);
  int*   posr  = (int*)  alloc((size_t)2 * T_TOK * 4);
  int*   list  = (int*)  alloc((size_t)2 * T_TOK * 4);
  int*   misc  = (int*)  alloc(64 * 4);
  int* cnt = misc, *cnt2 = misc + 8, *offs = misc + 16;

  k_embed<<<(T_TOK * F_DIM) / 256, 256, 0, stream>>>(inp, embed, x0, x1, x0b, x1b);
  k_cvt<<<(N_CLS * 2 * F_DIM) / 256, 256, 0, stream>>>(out_w, outwb);

  for (int d = 0; d < D_LAY; ++d) {
    // ---- weight bf16 conversion for this layer ----
    k_cvt<<<(E_EXP * F_DIM * I_DIM) / 256, 256, 0, stream>>>(
        w_in + (size_t)d * E_EXP * F_DIM * I_DIM, winb);
    k_cvt<<<(E_EXP * I_DIM * F3) / 256, 256, 0, stream>>>(
        w_out + (size_t)d * E_EXP * I_DIM * F3, woutb);
    k_convw<<<(K_TAP * I_DIM * I_DIM) / 256, 256, 0, stream>>>(
        conv_w + (size_t)d * I_DIM * I_DIM * K_TAP, cwt);

    // ---- MoE in: h = relu(top2-combine(x1 @ w_in[e])) ----
    k_gate<<<T_TOK / 256, 256, 0, stream>>>(x1, F_DIM, g_in_w + (size_t)d * F_DIM * E_EXP, eidx, esc);
    k_zero64<<<1, 64, 0, stream>>>(misc);
    k_count<<<T_TOK / 256, 256, 0, stream>>>(eidx, cnt);
    k_scan<<<1, 1, 0, stream>>>(cnt, offs);
    k_bucket<<<T_TOK / 256, 256, 0, stream>>>(eidx, offs, cnt2, list, posr);
    gemm_bf16<A_GATHER, B_BF16ROW><<<dim3(I_DIM / TBN, T_TOK / TBM, E_EXP), 256, 0, stream>>>(
        x1b, F_DIM, T_TOK, F_DIM,
        winb, nullptr, I_DIM, 0,
        Y, I_DIM, nullptr, 0, S_SEQ,
        list, offs, cnt, (long)F_DIM * I_DIM,
        nullptr, 0, 0);
    k_combine_bf<<<(T_TOK * I_DIM) / 256, 256, 0, stream>>>(Y, I_DIM, posr, esc, h_bf);

    // ---- causal conv as GEMM over K = I*K_TAP, hc = relu(conv(h)) ----
    gemm_bf16<A_DIRECT, B_BF16ROW><<<dim3(I_DIM / TBN, T_TOK / TBM, 1), 256, 0, stream>>>(
        h_bf, I_DIM, T_TOK, I_DIM * K_TAP,
        cwt, nullptr, I_DIM, 0,
        hc, I_DIM, hc_bf, /*convI=*/I_DIM, S_SEQ,
        nullptr, nullptr, nullptr, 0,
        nullptr, /*relu=*/1, 0);

    // ---- MoE out: o = top2-combine(hc @ w_out[e]) ----
    k_gate<<<T_TOK / 256, 256, 0, stream>>>(hc, I_DIM, g_out_w + (size_t)d * I_DIM * E_EXP, eidx, esc);
    k_zero64<<<1, 64, 0, stream>>>(misc);
    k_count<<<T_TOK / 256, 256, 0, stream>>>(eidx, cnt);
    k_scan<<<1, 1, 0, stream>>>(cnt, offs);
    k_bucket<<<T_TOK / 256, 256, 0, stream>>>(eidx, offs, cnt2, list, posr);
    gemm_bf16<A_GATHER, B_BF16ROW><<<dim3(F3 / TBN, T_TOK / TBM, E_EXP), 256, 0, stream>>>(
        hc_bf, I_DIM, T_TOK, I_DIM,
        woutb, nullptr, F3, 0,
        Y, F3, nullptr, 0, S_SEQ,
        list, offs, cnt, (long)I_DIM * F3,
        nullptr, 0, 0);
    k_combine_f32<<<(T_TOK * F3) / 256, 256, 0, stream>>>(Y, F3, posr, esc, obuf);

    // ---- cumsum/divisor*scale+shift, norm, residual recurrence ----
    k_cumsum<<<(B_BATCH * F_DIM) / 256, 256, 0, stream>>>(obuf, g);
    k_normupd<<<T_TOK, 256, 0, stream>>>(g, x0, x1, x0b, x1b);
  }

  // ---- final logits: M = classes (A = out_w bf16 row-major), N = tokens
  // (B transposed from token-major x0b|x1b), output written (B, CLASSES, S) ----
  gemm_bf16<A_DIRECT, B_XPOSE><<<dim3(T_TOK / TBN, N_CLS / TBM, 1), 256, 0, stream>>>(
      outwb, 2 * F_DIM, N_CLS, 2 * F_DIM,
      x0b, x1b, F_DIM, /*bsplit=*/F_DIM,
      logits, S_SEQ, nullptr, 0, S_SEQ,
      nullptr, nullptr, nullptr, 0,
      out_b, 0, (long)N_CLS * S_SEQ);
}
